// SelfAttention_4063039062647
// MI455X (gfx1250) — matrix operand order repaired and checked by reading, not compiled, not run on hardware
//
#include <hip/hip_runtime.h>

// ---------------------------------------------------------------------------
// GQA self-attention block for MI455X (gfx1250, wave32, WMMA).
//   q = x@Wq+bq ; k = x@Wk+bk ; v = x@Wv+bv
//   scores = (q.kT / 8) * mask ; w = softmax ; o = w@v ; out = o@Wo+bo+x
// All matmuls via v_wmma_f32_16x16x32_f16 (f16 in, f32 acc).
// Flash-style attention (online softmax, base-2 domain) -> no [B,NH,S,S]
// score tensor (256MB of HBM traffic avoided).
// GEMM: double-buffered LDS, A-tile staged with CDNA5 async-to-LDS DMA
// (global_load_async_to_lds_b128, ASYNCcnt), B-tile transposed through regs.
// ---------------------------------------------------------------------------

#define HIDS 2048
#define NHS  32
#define GS   8
#define RS   4      // heads per kv group
#define HDS  64

typedef _Float16 h16;
typedef __attribute__((ext_vector_type(16))) _Float16 v16h;
typedef __attribute__((ext_vector_type(8)))  _Float16 v8h;
typedef __attribute__((ext_vector_type(8)))  float    v8f;

union FragH { v16h v; unsigned u[8]; };

// ---- CDNA5 async copy: global -> LDS, 16B per lane, tracked on ASYNCcnt ----
__device__ __forceinline__ void async_copy_b128(const h16* gsrc, h16* ldst) {
  unsigned           loff = (unsigned)(uintptr_t)ldst;           // LDS offset
  unsigned long long gadr = (unsigned long long)(uintptr_t)gsrc;
  asm volatile("global_load_async_to_lds_b128 %0, %1, off"
               :: "v"(loff), "v"(gadr) : "memory");
}
__device__ __forceinline__ void wait_async0() {
  asm volatile("s_wait_asynccnt 0" ::: "memory");
}

// ---------------------------------------------------------------------------
// fp32 -> fp16 conversion, 8 elements / thread
// ---------------------------------------------------------------------------
__global__ __launch_bounds__(256) void cvt_f32_f16(const float* __restrict__ in,
                                                   h16* __restrict__ out, int n8) {
  int i = blockIdx.x * blockDim.x + threadIdx.x;   // index in units of 8
  if (i >= n8) return;
  float4 a = ((const float4*)in)[i * 2];
  float4 b = ((const float4*)in)[i * 2 + 1];
  v8h r = { (h16)a.x, (h16)a.y, (h16)a.z, (h16)a.w,
            (h16)b.x, (h16)b.y, (h16)b.z, (h16)b.w };
  *(v8h*)(out + (size_t)i * 8) = r;
}

// ---------------------------------------------------------------------------
// Tiled WMMA GEMM:  out[M,N] = A[M,K] @ W[K,N] + bias (+resid if FINAL)
// Block tile 128x128, 8 waves (4 along M x 2 along N), wave tile 32x64,
// K-step 32, double-buffered LDS, one barrier per K-step.
//   A tile [m][k]: async DMA to LDS (contiguous copy).
//   W tile -> lsB[n][k] (transposed): global->regs early, regs->LDS late.
// FINAL=false: f16 out, no residual. FINAL=true: fp32 out + residual.
// Requires M%128==0, N%128==0, K%32==0.
// ---------------------------------------------------------------------------
template <bool FINAL>
__global__ __launch_bounds__(256) void gemm_f16_wmma(
    const h16* __restrict__ A, const h16* __restrict__ Wt,
    const float* __restrict__ bias, const float* __restrict__ resid,
    h16* __restrict__ outH, float* __restrict__ outF,
    int M, int N, int K)
{
  __shared__ __align__(16) h16 lsA[2][128][32];   // [m][k]  2 x 8KB
  __shared__ __align__(16) h16 lsB[2][128][32];   // [n][k]  2 x 8KB

  const int tid   = threadIdx.x;
  const int lane  = tid & 31;
  const int w     = tid >> 5;        // wave 0..7
  const int wm    = w & 3;           // wave row (M)
  const int wn    = w >> 2;          // wave col (N)
  const int bm    = blockIdx.y * 128;
  const int bn    = blockIdx.x * 128;
  const int lhalf = lane & 15;
  const int hi    = lane >> 4;

  // per-thread staging coordinates
  const int arow = tid >> 1;             // A: 0..127
  const int acol = (tid & 1) * 16;       // A: 0 or 16
  const int kB   = tid & 31;             // B: k within tile
  const int n0B  = (tid >> 5) * 16;      // B: n start (16 cols)

  const h16* Abase = A  + (size_t)(bm + arow) * K + acol;
  const h16* Wbase = Wt + (size_t)kB * N + bn + n0B;
  const size_t WkStride = (size_t)32 * N;

  v8f acc[2][4];
  #pragma unroll
  for (int i = 0; i < 2; i++)
    #pragma unroll
    for (int j = 0; j < 4; j++)
      #pragma unroll
      for (int e = 0; e < 8; e++) acc[i][j][e] = 0.0f;

  // ---- prologue: stage tile 0 into buffer 0 ----
  async_copy_b128(Abase,     &lsA[0][arow][acol]);
  async_copy_b128(Abase + 8, &lsA[0][arow][acol + 8]);
  {
    v8h w0 = *(const v8h*)Wbase;
    v8h w1 = *(const v8h*)(Wbase + 8);
    #pragma unroll
    for (int i = 0; i < 8; i++) {
      lsB[0][n0B + i][kB]     = w0[i];
      lsB[0][n0B + 8 + i][kB] = w1[i];
    }
  }
  wait_async0();
  __syncthreads();

  int p = 0;
  for (int kk = 0; kk < K; kk += 32) {
    const bool has_next = (kk + 32) < K;

    // ---- issue async A-tile DMA for next K-step into buffer p^1 ----
    if (has_next) {
      const h16* an = Abase + kk + 32;
      async_copy_b128(an,     &lsA[p ^ 1][arow][acol]);
      async_copy_b128(an + 8, &lsA[p ^ 1][arow][acol + 8]);
    }
    // ---- issue global loads for next W tile (consumed after compute) ----
    v8h nw0, nw1;
    if (has_next) {
      const h16* wn_ = Wbase + (size_t)(kk + 32) * N;
      nw0 = *(const v8h*)wn_;
      nw1 = *(const v8h*)(wn_ + 8);
    }

    // ---- fragments from current buffers (ISA 7.12.2 layouts) ----
    FragH afr[2], bfr[4];
    #pragma unroll
    for (int mt = 0; mt < 2; mt++) {
      int row = wm * 32 + mt * 16 + lhalf;          // A: M = lane&15
      #pragma unroll
      for (int v = 0; v < 8; v++) {
        int k0 = ((v & 4) ? 16 : 0) + hi * 8 + (v & 3) * 2;
        afr[mt].u[v] = *(const unsigned*)&lsA[p][row][k0];
      }
    }
    #pragma unroll
    for (int nt = 0; nt < 4; nt++) {
      int col = wn * 64 + nt * 16 + lhalf;          // B: N = lane&15
      #pragma unroll
      for (int v = 0; v < 8; v++) {
        int k0 = ((v & 4) ? 16 : 0) + hi * 8 + (v & 3) * 2;  // B: same K order as A
        bfr[nt].u[v] = *(const unsigned*)&lsB[p][col][k0];
      }
    }

    #pragma unroll
    for (int mt = 0; mt < 2; mt++)
      #pragma unroll
      for (int nt = 0; nt < 4; nt++)
        acc[mt][nt] = __builtin_amdgcn_wmma_f32_16x16x32_f16(
            false, afr[mt].v, false, bfr[nt].v, (short)0, acc[mt][nt],
            false, false);

    // ---- commit next W tile (transposed) into buffer p^1 ----
    if (has_next) {
      #pragma unroll
      for (int i = 0; i < 8; i++) {
        lsB[p ^ 1][n0B + i][kB]     = nw0[i];
        lsB[p ^ 1][n0B + 8 + i][kB] = nw1[i];
      }
      wait_async0();
    }
    __syncthreads();
    p ^= 1;
  }

  // ---- epilogue (compile-time specialized, branch-free) ----
  #pragma unroll
  for (int mt = 0; mt < 2; mt++) {
    #pragma unroll
    for (int v = 0; v < 8; v++) {
      const int mrow = bm + wm * 32 + mt * 16 + v + hi * 8;  // C/D: row=v+hi*8
      const size_t rowoff = (size_t)mrow * N;
      #pragma unroll
      for (int nt = 0; nt < 4; nt++) {
        const int n = bn + wn * 64 + nt * 16 + lhalf;
        float val = acc[mt][nt][v] + bias[n];
        if constexpr (FINAL) {
          val += resid[rowoff + n];
          outF[rowoff + n] = val;
        } else {
          outH[rowoff + n] = (h16)val;
        }
      }
    }
  }
}

// ---------------------------------------------------------------------------
// Flash attention (GQA). One block per (64-query tile, b*NH head).
// 4 waves x 16 query rows. Key tiles of 32. Multiplicative mask on scaled
// logits; online softmax kept in base-2 domain (v_exp_f32 is exp2 natively).
// ---------------------------------------------------------------------------
__global__ __launch_bounds__(128) void attn_wmma(
    const h16* __restrict__ Q, const h16* __restrict__ Kh,
    const h16* __restrict__ Vh, const float* __restrict__ mask,
    h16* __restrict__ Oh, int B, int S)
{
  __shared__ __align__(16) h16   lsK[32][HDS];    // key tile     4KB
  __shared__ __align__(16) h16   lsV[32][HDS];    // value tile   4KB
  __shared__ __align__(16) float lsM[64][32];     // mask tile    8KB
  __shared__ __align__(16) h16   lsP[4][16][32];  // per-wave P   4KB

  const int tid   = threadIdx.x;
  const int lane  = tid & 31;
  const int w     = tid >> 5;
  const int lhalf = lane & 15;
  const int hi    = lane >> 4;
  const int KV    = GS * HDS;

  const int headId = blockIdx.y;        // 0 .. B*NH-1
  const int b  = headId / NHS;
  const int h  = headId % NHS;          // q head; kv group = h / R
  const int g  = h / RS;
  const int qbase = blockIdx.x * 64;
  const int qrow  = qbase + w * 16;     // this wave's 16 query rows

  // staging coordinates
  const int krow = tid >> 2, kcol = (tid & 3) * 16;   // K/V tile 32x64
  const int mrow = tid >> 1, mcol = (tid & 1) * 16;   // mask tile 64x32
  const h16*   Kbase = Kh + ((size_t)b * S + krow) * KV + g * HDS + kcol;
  const h16*   Vbase = Vh + ((size_t)b * S + krow) * KV + g * HDS + kcol;
  const float* Mbase = mask + ((size_t)b * S + qbase + mrow) * S + mcol;

  // ---- Q A-fragments for this wave: 16 x 64 -> two 16x32 frags ----
  FragH qa[2];
  {
    const h16* qp = Q + ((size_t)b * S + qrow + lhalf) * HIDS + h * HDS;
    #pragma unroll
    for (int kh = 0; kh < 2; kh++)
      #pragma unroll
      for (int v = 0; v < 8; v++) {
        int k0 = kh * 32 + ((v & 4) ? 16 : 0) + hi * 8 + (v & 3) * 2;
        qa[kh].u[v] = *(const unsigned*)(qp + k0);
      }
  }

  float rm[8], rl[8];                   // per-row running max / sum (log2 dom)
  v8f o[4];                             // 16x64 output accumulator
  #pragma unroll
  for (int v = 0; v < 8; v++) { rm[v] = -1e30f; rl[v] = 0.0f; }
  #pragma unroll
  for (int nt = 0; nt < 4; nt++)
    #pragma unroll
    for (int e = 0; e < 8; e++) o[nt][e] = 0.0f;

  // (1/sqrt(64)) * log2(e): logits scaled straight into base-2 domain
  const float c_scale = 0.125f * 1.44269504088896f;

  for (int kt = 0; kt < S; kt += 32) {
    // ---- prefetch next tiles into cache (global_prefetch_b8) ----
    if (kt + 32 < S) {
      __builtin_prefetch(Kbase + (size_t)(kt + 32) * KV, 0, 1);
      __builtin_prefetch(Vbase + (size_t)(kt + 32) * KV, 0, 1);
      __builtin_prefetch(Mbase + kt + 32, 0, 1);
    }
    // ---- cooperative staging of K, V (32x64 f16) and mask (64x32 f32) ----
    {
      const h16* kp = Kbase + (size_t)kt * KV;
      const h16* vp = Vbase + (size_t)kt * KV;
      *(v8h*)&lsK[krow][kcol]     = *(const v8h*)kp;
      *(v8h*)&lsK[krow][kcol + 8] = *(const v8h*)(kp + 8);
      *(v8h*)&lsV[krow][kcol]     = *(const v8h*)vp;
      *(v8h*)&lsV[krow][kcol + 8] = *(const v8h*)(vp + 8);
      const float* mp = Mbase + kt;
      #pragma unroll
      for (int i = 0; i < 16; i += 4)
        *(float4*)&lsM[mrow][mcol + i] = *(const float4*)(mp + i);
    }
    __syncthreads();

    // ---- scores: 16 rows x 32 keys = two 16x16 C frags ----
    v8f sc[2];
    #pragma unroll
    for (int f = 0; f < 2; f++)
      #pragma unroll
      for (int e = 0; e < 8; e++) sc[f][e] = 0.0f;

    #pragma unroll
    for (int f = 0; f < 2; f++) {
      int key = f * 16 + lhalf;               // B: N(=key) = lane&15
      #pragma unroll
      for (int kh = 0; kh < 2; kh++) {
        FragH bk;                              // B[kd][key] = K[key][kd]
        #pragma unroll
        for (int v = 0; v < 8; v++) {
          int kd = kh * 32 + ((v & 4) ? 16 : 0) + hi * 8 + (v & 3) * 2;  // hd pair, A-frag K order
          bk.u[v] = *(const unsigned*)&lsK[key][kd];
        }
        sc[f] = __builtin_amdgcn_wmma_f32_16x16x32_f16(
            false, qa[kh].v, false, bk.v, (short)0, sc[f], false, false);
      }
    }

    // ---- scale + multiplicative mask (already in log2 domain) ----
    float pv[2][8];
    #pragma unroll
    for (int f = 0; f < 2; f++)
      #pragma unroll
      for (int v = 0; v < 8; v++) {
        int rloc = w * 16 + v + hi * 8;
        pv[f][v] = sc[f][v] * lsM[rloc][f * 16 + lhalf] * c_scale;
      }

    // ---- online softmax (row stats live in 16-lane N groups) ----
    #pragma unroll
    for (int v = 0; v < 8; v++) {
      float t = fmaxf(pv[0][v], pv[1][v]);
      #pragma unroll
      for (int m = 1; m < 16; m <<= 1) t = fmaxf(t, __shfl_xor(t, m, 32));
      float mnew  = fmaxf(rm[v], t);
      float alpha = exp2f(rm[v] - mnew);
      pv[0][v] = exp2f(pv[0][v] - mnew);
      pv[1][v] = exp2f(pv[1][v] - mnew);
      float rs = pv[0][v] + pv[1][v];
      #pragma unroll
      for (int m = 1; m < 16; m <<= 1) rs += __shfl_xor(rs, m, 32);
      rl[v] = rl[v] * alpha + rs;
      rm[v] = mnew;
      #pragma unroll
      for (int nt = 0; nt < 4; nt++) o[nt][v] *= alpha;
    }

    // ---- transpose P (C layout -> A layout) through wave-private LDS ----
    #pragma unroll
    for (int f = 0; f < 2; f++)
      #pragma unroll
      for (int v = 0; v < 8; v++)
        lsP[w][v + hi * 8][f * 16 + lhalf] = (h16)pv[f][v];
    // DS ops are per-wave in-order; only this wave's stores must land.
    asm volatile("s_wait_dscnt 0" ::: "memory");

    FragH pa;
    #pragma unroll
    for (int v = 0; v < 8; v++) {
      int k0 = ((v & 4) ? 16 : 0) + hi * 8 + (v & 3) * 2;
      pa.u[v] = *(const unsigned*)&lsP[w][lhalf][k0];
    }

    // ---- O += P @ V ----
    #pragma unroll
    for (int nt = 0; nt < 4; nt++) {
      FragH bv;
      int n = nt * 16 + lhalf;                 // B: N(=hd) = lane&15
      #pragma unroll
      for (int v = 0; v < 8; v++) {
        int k0 = ((v & 4) ? 16 : 0) + hi * 8 + (v & 3) * 2;  // key index pair (stride HD)
        unsigned lo = *(const unsigned short*)&lsV[k0][n];
        unsigned hw = *(const unsigned short*)&lsV[k0 + 1][n];
        bv.u[v] = lo | (hw << 16);
      }
      o[nt] = __builtin_amdgcn_wmma_f32_16x16x32_f16(
          false, pa.v, false, bv.v, (short)0, o[nt], false, false);
    }
    __syncthreads();   // protect lsK/lsV/lsM before next tile's staging
  }

  // ---- normalize and store f16 for the O-projection GEMM ----
  #pragma unroll
  for (int v = 0; v < 8; v++) rl[v] = 1.0f / rl[v];
  #pragma unroll
  for (int nt = 0; nt < 4; nt++)
    #pragma unroll
    for (int v = 0; v < 8; v++) {
      int rq = qrow + v + hi * 8;
      float val = o[nt][v] * rl[v];
      Oh[((size_t)b * S + rq) * HIDS + h * HDS + nt * 16 + lhalf] = (h16)val;
    }
}

// ---------------------------------------------------------------------------
// Host-side orchestration
// inputs: x, attention_mask, Wq, bq, Wk, bk, Wv, bv, Wo, bo
// ---------------------------------------------------------------------------
extern "C" void kernel_launch(void* const* d_in, const int* in_sizes, int n_in,
                              void* d_out, int out_size, void* d_ws, size_t ws_size,
                              hipStream_t stream) {
  const float* x    = (const float*)d_in[0];
  const float* mask = (const float*)d_in[1];
  const float* Wq   = (const float*)d_in[2];
  const float* bq   = (const float*)d_in[3];
  const float* Wk   = (const float*)d_in[4];
  const float* bk   = (const float*)d_in[5];
  const float* Wv   = (const float*)d_in[6];
  const float* bv   = (const float*)d_in[7];
  const float* Wo   = (const float*)d_in[8];
  const float* bo   = (const float*)d_in[9];

  const int BS = in_sizes[0] / HIDS;        // B*S
  const int S  = in_sizes[1] / BS;          // mask is B*S*S
  const int B  = BS / S;
  const int KV = GS * HDS;                  // 512

  // f16 workspace partition (~40MB used)
  h16* xh  = (h16*)d_ws;
  h16* Wqh = xh  + (size_t)BS * HIDS;
  h16* Wkh = Wqh + (size_t)HIDS * HIDS;
  h16* Wvh = Wkh + (size_t)HIDS * KV;
  h16* Woh = Wvh + (size_t)HIDS * KV;
  h16* Qh  = Woh + (size_t)HIDS * HIDS;
  h16* Kh  = Qh  + (size_t)BS * HIDS;
  h16* Vh  = Kh  + (size_t)BS * KV;
  h16* Ah  = xh;   // alias: xh is dead after the V projection

  // 1) fp32 -> fp16 conversions (8 elems/thread)
  {
    size_t n;
    n = (size_t)BS * HIDS / 8;   cvt_f32_f16<<<(n + 255) / 256, 256, 0, stream>>>(x,  xh,  (int)n);
    n = (size_t)HIDS * HIDS / 8; cvt_f32_f16<<<(n + 255) / 256, 256, 0, stream>>>(Wq, Wqh, (int)n);
    n = (size_t)HIDS * KV / 8;   cvt_f32_f16<<<(n + 255) / 256, 256, 0, stream>>>(Wk, Wkh, (int)n);
    n = (size_t)HIDS * KV / 8;   cvt_f32_f16<<<(n + 255) / 256, 256, 0, stream>>>(Wv, Wvh, (int)n);
    n = (size_t)HIDS * HIDS / 8; cvt_f32_f16<<<(n + 255) / 256, 256, 0, stream>>>(Wo, Woh, (int)n);
  }

  // 2) Q/K/V projections (WMMA, f16 out, bias fused)
  gemm_f16_wmma<false><<<dim3(HIDS / 128, BS / 128), 256, 0, stream>>>(
      xh, Wqh, bq, nullptr, Qh, nullptr, BS, HIDS, HIDS);
  gemm_f16_wmma<false><<<dim3(KV / 128, BS / 128), 256, 0, stream>>>(
      xh, Wkh, bk, nullptr, Kh, nullptr, BS, KV, HIDS);
  gemm_f16_wmma<false><<<dim3(KV / 128, BS / 128), 256, 0, stream>>>(
      xh, Wvh, bv, nullptr, Vh, nullptr, BS, KV, HIDS);

  // 3) flash attention (writes Ah, aliasing xh)
  attn_wmma<<<dim3(S / 64, B * NHS), 128, 0, stream>>>(Qh, Kh, Vh, mask, Ah, B, S);

  // 4) O-projection + bias + residual -> fp32 d_out
  gemm_f16_wmma<true><<<dim3(HIDS / 128, BS / 128), 256, 0, stream>>>(
      Ah, Woh, bo, x, nullptr, (float*)d_out, BS, HIDS, HIDS);
}
